// LocalPPFTransformer_70987219469000
// MI455X (gfx1250) — matrix operand order repaired and checked by reading, not compiled, not run on hardware
//
#include <hip/hip_runtime.h>
#include <hip/hip_bf16.h>

typedef __attribute__((ext_vector_type(16))) _Float16 v16h;
typedef __attribute__((ext_vector_type(8)))  _Float16 v8h;
typedef __attribute__((ext_vector_type(8)))  float    v8f;
typedef __attribute__((ext_vector_type(4)))  unsigned int u32x4;
typedef __attribute__((ext_vector_type(8)))  int i32x8;
typedef __attribute__((ext_vector_type(4)))  int i32x4;

#define WMMA16(a,b,c) __builtin_amdgcn_wmma_f32_16x16x32_f16(false,(a),false,(b),(short)0,(c),false,false)

__device__ __forceinline__ void lds_fence() {
  asm volatile("s_wait_dscnt 0" ::: "memory");
}

// ---- Tensor Data Mover: DMA a contiguous block (bytes % 8 == 0, bytes/8 <= 65535)
// from global memory into LDS. D# built per CDNA5 ISA 8.3/8.4 (1-D tile,
// data_size=8B, count=1, type=2). Issued once per calling wave; tracked by TENSORcnt.
// This toolchain's builtin arity is 6: (g0, g1, g2, g3, g4, cpol).
__device__ __forceinline__ void tdm_load_to_lds(const void* gsrc, unsigned lds_off, unsigned bytes) {
  unsigned long long ga = (unsigned long long)(uintptr_t)gsrc;
  unsigned n8 = bytes >> 3;  // elements of 8 bytes
  u32x4 g0;
  g0.x = 1u;                                                 // count=1, no gather
  g0.y = lds_off;                                            // lds_addr [63:32]
  g0.z = (unsigned)(ga & 0xffffffffu);                       // global_addr [95:64]
  g0.w = (unsigned)((ga >> 32) & 0x1ffffffu) | (2u << 30);   // global_addr[56:32] | type=2
  i32x8 g1;
  g1[0] = 0x00030000;                           // wg_mask=0, data_size=3 (8B)
  g1[1] = (int)((n8 & 0xffffu) << 16);          // tensor_dim0[15:0] @ bits 63:48
  g1[2] = (int)((n8 >> 16) & 0xffffu) | (1 << 16); // tensor_dim0[31:16] | tensor_dim1=1
  g1[3] = (int)((n8 & 0xffffu) << 16);          // tile_dim0 = n8 @ bits 127:112
  g1[4] = 1;                                    // tile_dim1=1, tile_dim2=0
  g1[5] = (int)n8;                              // tensor_dim0_stride[31:0]
  g1[6] = 0;                                    // stride hi, tensor_dim1_stride lo
  g1[7] = 0;
  i32x4 z4 = {0, 0, 0, 0};
  i32x8 z8 = {0, 0, 0, 0, 0, 0, 0, 0};
  __builtin_amdgcn_tensor_load_to_lds(g0, g1, z4, z4, z8, 0);
}

__device__ __forceinline__ void tdm_wait() {
  __builtin_amdgcn_s_wait_tensorcnt(0);
}

union HF16 { v16h v; v8h h[2]; _Float16 e[16]; };

// A fragment (16x32 f16, row-major source with given stride in halves).
// lanes 0-15: M=lane, K = kb + {0..7, 16..23}; lanes 16-31: M=lane-16, K = kb + {8..15, 24..31}.
__device__ __forceinline__ v16h load_a_f16(const _Float16* src, int stride, int kb,
                                           int l16, int hiL) {
  const _Float16* p = src + (size_t)l16 * stride + kb + hiL * 8;
  HF16 u;
  u.h[0] = *(const v8h*)(p);
  u.h[1] = *(const v8h*)(p + 16);
  return u.v;
}

// B fragment (32x16 f16) from a column-major weight [N][Kdim]:
// lanes 0-15: N=lane, K = kb + {0..7, 16..23}; lanes 16-31: N=lane-16, K = kb + {8..15, 24..31}.
__device__ __forceinline__ v16h load_b_cm(const _Float16* Wcm, int kdim, int nb, int kb,
                                          int l16, int hiL) {
  const _Float16* p = Wcm + (size_t)(nb + l16) * kdim + kb + hiL * 8;
  HF16 u;
  u.h[0] = *(const v8h*)(p);
  u.h[1] = *(const v8h*)(p + 16);
  return u.v;
}

// A fragment converted on the fly from an f32 row-major source.
__device__ __forceinline__ v16h load_a_f32cvt(const float* src, int stride, int kb,
                                              int l16, int hiL) {
  const float* p = src + (size_t)l16 * stride + kb + hiL * 8;
  HF16 u;
#pragma unroll
  for (int i = 0; i < 8; ++i) { u.e[i] = (_Float16)p[i]; u.e[8 + i] = (_Float16)p[16 + i]; }
  return u.v;
}

// ---------------- weight convert: f32 row-major [K][N] -> f16 col-major [N][K]
__global__ void cvt_w(const float* __restrict__ src, _Float16* __restrict__ dst, int K, int N) {
  int i = blockIdx.x * blockDim.x + threadIdx.x;
  if (i < K * N) {
    int k = i / N, n = i % N;
    dst[(size_t)n * K + k] = (_Float16)src[i];
  }
}

// ---------------- K1: f = feats @ W_in + b_in  (N x 64) x (64 x 128)
__global__ void k_proj_in(const float* __restrict__ feats, const _Float16* __restrict__ Win_cm,
                          const float* __restrict__ b_in, float* __restrict__ fbuf,
                          _Float16* __restrict__ f16buf, int Nrows) {
  extern __shared__ char smem[];
  _Float16* Wl = (_Float16*)smem;  // 128x64 col-major (16KB)
  if ((threadIdx.x >> 5) == 0) {
    tdm_load_to_lds(Win_cm, (unsigned)(uintptr_t)Wl, 128 * 64 * 2);
    tdm_wait();
  }
  __syncthreads();

  int wave = threadIdx.x >> 5;
  int tile = blockIdx.x * (blockDim.x >> 5) + wave;
  if (tile * 16 >= Nrows) return;
  int lane = threadIdx.x & 31, l16 = lane & 15, hiL = lane >> 4;

  v8f acc[8] = {};
  const float* A = feats + (size_t)tile * 16 * 64;
#pragma unroll
  for (int c = 0; c < 2; ++c) {
    v16h a = load_a_f32cvt(A, 64, c * 32, l16, hiL);
#pragma unroll
    for (int n = 0; n < 8; ++n) {
      v16h b = load_b_cm(Wl, 64, n * 16, c * 32, l16, hiL);
      acc[n] = WMMA16(a, b, acc[n]);
    }
  }
#pragma unroll
  for (int n = 0; n < 8; ++n) {
    int col = n * 16 + l16;
    float bias = b_in[col];
#pragma unroll
    for (int v = 0; v < 8; ++v) {
      int row = tile * 16 + hiL * 8 + v;
      float val = acc[n][v] + bias;
      fbuf[(size_t)row * 128 + col] = val;
      f16buf[(size_t)row * 128 + col] = (_Float16)val;
    }
  }
}

// ---------------- K2: q/k/v = f @ W_{q,k,v} + b, stored f16 (weights in LDS via TDM)
__global__ void k_proj_qkv(const _Float16* __restrict__ f16buf,
                           const _Float16* Wq, const _Float16* Wk, const _Float16* Wv,
                           const float* bq, const float* bk, const float* bv,
                           _Float16* qf, _Float16* kf, _Float16* vf, int Nrows) {
  extern __shared__ char smem[];
  _Float16* Wl0 = (_Float16*)smem;            // 32KB each
  _Float16* Wl1 = Wl0 + 128 * 128;
  _Float16* Wl2 = Wl1 + 128 * 128;
  if ((threadIdx.x >> 5) == 0) {
    tdm_load_to_lds(Wq, (unsigned)(uintptr_t)Wl0, 128 * 128 * 2);
    tdm_load_to_lds(Wk, (unsigned)(uintptr_t)Wl1, 128 * 128 * 2);
    tdm_load_to_lds(Wv, (unsigned)(uintptr_t)Wl2, 128 * 128 * 2);
    tdm_wait();
  }
  __syncthreads();

  int wave = threadIdx.x >> 5;
  int tile = blockIdx.x * (blockDim.x >> 5) + wave;
  if (tile * 16 >= Nrows) return;
  int lane = threadIdx.x & 31, l16 = lane & 15, hiL = lane >> 4;

  const _Float16* Ws[3] = {Wl0, Wl1, Wl2};
  const float* bs[3] = {bq, bk, bv};
  _Float16* outs[3] = {qf, kf, vf};
  const _Float16* A = f16buf + (size_t)tile * 16 * 128;

  for (int w = 0; w < 3; ++w) {
    v8f acc[8] = {};
#pragma unroll
    for (int c = 0; c < 4; ++c) {
      v16h a = load_a_f16(A, 128, c * 32, l16, hiL);
#pragma unroll
      for (int n = 0; n < 8; ++n) {
        v16h b = load_b_cm(Ws[w], 128, n * 16, c * 32, l16, hiL);
        acc[n] = WMMA16(a, b, acc[n]);
      }
    }
#pragma unroll
    for (int n = 0; n < 8; ++n) {
      int col = n * 16 + l16;
      float bias = bs[w][col];
#pragma unroll
      for (int v = 0; v < 8; ++v) {
        int row = tile * 16 + hiL * 8 + v;
        outs[w][(size_t)row * 128 + col] = (_Float16)(acc[n][v] + bias);
      }
    }
  }
}

// ---------------- K3: fused PPF embedding + W_a(+max) + W_p + attention -> o16
#define FACTOR_A 3.8197186342054885f   // 180/(15*pi)
#define DIVC    (-0.14391156831f)      // -ln(10000)/64
// per-wave scratch: [0,4096) pos f16 ALIASED under [0,8192) p f32 (phase-ordered,
// per-wave LDS is in-order), + 256B scores + 256B attn = 8704B
#define SCR3_PER_WAVE 8704

__global__ void k_ppf(const float* __restrict__ ppfs, const int* __restrict__ node_idx,
                      const int* __restrict__ group_idx,
                      const _Float16* Wa_cm, const float* b_a,
                      const _Float16* Wp_cm, const float* b_p,
                      const _Float16* qf, const _Float16* kf, const _Float16* vf,
                      _Float16* __restrict__ o16, int M) {
  extern __shared__ char smem[];
  _Float16* Wa = (_Float16*)smem;             // 32KB
  _Float16* Wp = Wa + 128 * 128;              // 32KB
  char* scrbase = smem + 65536;
  if ((threadIdx.x >> 5) == 0) {
    tdm_load_to_lds(Wa_cm, (unsigned)(uintptr_t)Wa, 128 * 128 * 2);
    tdm_load_to_lds(Wp_cm, (unsigned)(uintptr_t)Wp, 128 * 128 * 2);
    tdm_wait();
  }
  __syncthreads();

  int wave = threadIdx.x >> 5;
  char* scr = scrbase + wave * SCR3_PER_WAVE;
  _Float16* posb = (_Float16*)scr;            // 16x128 f16 (phase 1-2)
  float* pb = (float*)scr;                    // 16x128 f32 (phase 3+, aliases posb)
  float* sb = (float*)(scr + 8192);           // 64 scores
  float* ab = sb + 64;                        // 64 attn

  int m = blockIdx.x * 4 + wave;
  if (m >= M) return;
  int lane = threadIdx.x & 31, l16 = lane & 15, hiL = lane >> 4;

  float ba[8], bp[8];
#pragma unroll
  for (int n = 0; n < 8; ++n) { ba[n] = b_a[n * 16 + l16]; bp[n] = b_p[n * 16 + l16]; }

  // ---- embedding GEMM with max over 4 angles ----
  v8f pacc[8];
#pragma unroll
  for (int n = 0; n < 8; ++n)
#pragma unroll
    for (int v = 0; v < 8; ++v) pacc[n][v] = -3.4e38f;

  for (int ang = 0; ang < 4; ++ang) {
    float x = ppfs[((size_t)m * 16 + l16) * 4 + ang] * FACTOR_A;
    v8f acc[8] = {};
#pragma unroll
    for (int c = 0; c < 4; ++c) {
      // build sin/cos A fragment directly in WMMA register layout
      HF16 ua;
#pragma unroll
      for (int v = 0; v < 8; ++v) {
        int base = c * 32 + hiL * 8 + ((v < 4) ? 2 * v : 16 + 2 * (v - 4));
        int j = base >> 1;
        float omega = x * __expf((float)j * DIVC);
        float s, cf;
        __sincosf(omega, &s, &cf);
        ua.e[2 * v] = (_Float16)s;
        ua.e[2 * v + 1] = (_Float16)cf;
      }
#pragma unroll
      for (int n = 0; n < 8; ++n) {
        v16h b = load_b_cm(Wa, 128, n * 16, c * 32, l16, hiL);
        acc[n] = WMMA16(ua.v, b, acc[n]);
      }
    }
#pragma unroll
    for (int n = 0; n < 8; ++n)
#pragma unroll
      for (int v = 0; v < 8; ++v)
        pacc[n][v] = fmaxf(pacc[n][v], acc[n][v] + ba[n]);
  }

  // C-layout -> LDS (f16) so we can re-read it in A-layout for the next GEMM
#pragma unroll
  for (int n = 0; n < 8; ++n)
#pragma unroll
    for (int v = 0; v < 8; ++v)
      posb[(hiL * 8 + v) * 128 + n * 16 + l16] = (_Float16)pacc[n][v];
  lds_fence();

  // ---- p = pos @ W_p + b_p ----
  v8f acc2[8] = {};
#pragma unroll
  for (int c = 0; c < 4; ++c) {
    v16h a = load_a_f16(posb, 128, c * 32, l16, hiL);
#pragma unroll
    for (int n = 0; n < 8; ++n) {
      v16h b = load_b_cm(Wp, 128, n * 16, c * 32, l16, hiL);
      acc2[n] = WMMA16(a, b, acc2[n]);
    }
  }
  // all posb reads issued before these writes; per-wave LDS is in-order
#pragma unroll
  for (int n = 0; n < 8; ++n)
#pragma unroll
    for (int v = 0; v < 8; ++v)
      pb[(hiL * 8 + v) * 128 + n * 16 + l16] = acc2[n][v] + bp[n];
  lds_fence();

  // ---- attention: lane handles neighbor kk=l16, two heads (2*hiL, 2*hiL+1) ----
  int qrow = node_idx[m];
  int g = group_idx[m * 16 + l16];
  const _Float16* qp = qf + (size_t)qrow * 128;
  const _Float16* kp = kf + (size_t)g * 128;
  int h0 = 2 * hiL;
  float s0 = 0.f, s1 = 0.f;
#pragma unroll 8
  for (int d = 0; d < 32; ++d) {
    s0 += (float)qp[h0 * 32 + d] * ((float)kp[h0 * 32 + d] + pb[l16 * 128 + h0 * 32 + d]);
    s1 += (float)qp[(h0 + 1) * 32 + d] * ((float)kp[(h0 + 1) * 32 + d] + pb[l16 * 128 + (h0 + 1) * 32 + d]);
  }
  const float scale = 0.17677669529663687f;  // 1/sqrt(32)
  sb[h0 * 16 + l16] = s0 * scale;
  sb[(h0 + 1) * 16 + l16] = s1 * scale;
  lds_fence();

  // softmax over 16 neighbors (redundant per lane, cheap)
#pragma unroll
  for (int hh = 0; hh < 2; ++hh) {
    int h = h0 + hh;
    float mx = -3.4e38f;
    for (int kk = 0; kk < 16; ++kk) mx = fmaxf(mx, sb[h * 16 + kk]);
    float den = 0.f;
    for (int kk = 0; kk < 16; ++kk) den += __expf(sb[h * 16 + kk] - mx);
    ab[h * 16 + l16] = __expf(sb[h * 16 + l16] - mx) / den;
  }
  lds_fence();

  // o[m][d] = sum_kk attn[h(d)][kk] * v[g[kk]][d]; lane covers d = t*32 + lane (head=t)
#pragma unroll
  for (int t = 0; t < 4; ++t) {
    int d = t * 32 + lane;
    float o = 0.f;
    for (int kk = 0; kk < 16; ++kk) {
      int g2 = group_idx[m * 16 + kk];
      o += ab[t * 16 + kk] * (float)vf[(size_t)g2 * 128 + d];
    }
    o16[(size_t)m * 128 + d] = (_Float16)o;
  }
}

// ---------------- K4: hidden = o@W_o + b_o; res = f[node]+hidden; LN; @W_out + b_out
#define SCR4_PER_WAVE 12288  // 8KB res f32 + 4KB normed f16 (disjoint: concurrent r/w)

__global__ void k_out(const _Float16* __restrict__ o16, const _Float16* Wo_cm, const float* b_o,
                      const float* __restrict__ fbuf, const int* __restrict__ node_idx,
                      const float* ln_g, const float* ln_b,
                      const _Float16* Wout_cm, const float* b_out,
                      float* __restrict__ out, int M) {
  extern __shared__ char smem[];
  _Float16* Wo = (_Float16*)smem;             // 32KB
  _Float16* Wout = Wo + 128 * 128;            // 32KB
  char* scrbase = smem + 65536;
  if ((threadIdx.x >> 5) == 0) {
    tdm_load_to_lds(Wo_cm, (unsigned)(uintptr_t)Wo, 128 * 128 * 2);
    tdm_load_to_lds(Wout_cm, (unsigned)(uintptr_t)Wout, 128 * 128 * 2);
    tdm_wait();
  }
  __syncthreads();

  int wave = threadIdx.x >> 5;
  float* rbuf = (float*)(scrbase + wave * SCR4_PER_WAVE);               // 16x128 f32
  _Float16* nbuf = (_Float16*)(scrbase + wave * SCR4_PER_WAVE + 8192);  // 16x128 f16

  int tile = blockIdx.x * 4 + wave;
  if (tile * 16 >= M) return;
  int m0 = tile * 16;
  int lane = threadIdx.x & 31, l16 = lane & 15, hiL = lane >> 4;

  // GEMM1: o @ W_o
  v8f acc[8] = {};
  const _Float16* A = o16 + (size_t)m0 * 128;
#pragma unroll
  for (int c = 0; c < 4; ++c) {
    v16h a = load_a_f16(A, 128, c * 32, l16, hiL);
#pragma unroll
    for (int n = 0; n < 8; ++n) {
      v16h b = load_b_cm(Wo, 128, n * 16, c * 32, l16, hiL);
      acc[n] = WMMA16(a, b, acc[n]);
    }
  }

  int nid[8];
#pragma unroll
  for (int v = 0; v < 8; ++v) nid[v] = node_idx[m0 + hiL * 8 + v];

#pragma unroll
  for (int n = 0; n < 8; ++n) {
    int col = n * 16 + l16;
    float bo = b_o[col];
#pragma unroll
    for (int v = 0; v < 8; ++v) {
      int r = hiL * 8 + v;
      rbuf[r * 128 + col] = acc[n][v] + bo + fbuf[(size_t)nid[v] * 128 + col];
    }
  }
  lds_fence();

  // LayerNorm on row l16 (lanes l16 and l16+16 redundantly compute stats)
  float sum = 0.f, sq = 0.f;
#pragma unroll 8
  for (int c = 0; c < 128; ++c) {
    float v2 = rbuf[l16 * 128 + c];
    sum += v2;
    sq += v2 * v2;
  }
  float mu = sum * (1.0f / 128.0f);
  float var = sq * (1.0f / 128.0f) - mu * mu;
  float rstd = rsqrtf(var + 1e-5f);
  int c0 = hiL * 64;
#pragma unroll 8
  for (int c = c0; c < c0 + 64; ++c) {
    float v2 = (rbuf[l16 * 128 + c] - mu) * rstd * ln_g[c] + ln_b[c];
    nbuf[l16 * 128 + c] = (_Float16)v2;
  }
  lds_fence();

  // GEMM2: normed @ W_out
  v8f acc2[8] = {};
#pragma unroll
  for (int c = 0; c < 4; ++c) {
    v16h a = load_a_f16(nbuf, 128, c * 32, l16, hiL);
#pragma unroll
    for (int n = 0; n < 8; ++n) {
      v16h b = load_b_cm(Wout, 128, n * 16, c * 32, l16, hiL);
      acc2[n] = WMMA16(a, b, acc2[n]);
    }
  }
#pragma unroll
  for (int n = 0; n < 8; ++n) {
    int col = n * 16 + l16;
    float bias = b_out[col];
#pragma unroll
    for (int v = 0; v < 8; ++v) {
      int row = m0 + hiL * 8 + v;
      out[(size_t)row * 128 + col] = acc2[n][v] + bias;
    }
  }
}

// ---------------- host launcher ----------------
extern "C" void kernel_launch(void* const* d_in, const int* in_sizes, int n_in,
                              void* d_out, int out_size, void* d_ws, size_t ws_size,
                              hipStream_t stream) {
  const float* feats = (const float*)d_in[0];
  const int* node_idx = (const int*)d_in[1];
  const int* group_idx = (const int*)d_in[2];
  const float* ppfs = (const float*)d_in[3];
  const float* W_a = (const float*)d_in[4];   const float* b_a = (const float*)d_in[5];
  const float* W_in = (const float*)d_in[6];  const float* b_in = (const float*)d_in[7];
  const float* W_q = (const float*)d_in[8];   const float* b_q = (const float*)d_in[9];
  const float* W_k = (const float*)d_in[10];  const float* b_k = (const float*)d_in[11];
  const float* W_v = (const float*)d_in[12];  const float* b_v = (const float*)d_in[13];
  const float* W_p = (const float*)d_in[14];  const float* b_p = (const float*)d_in[15];
  const float* W_o = (const float*)d_in[16];  const float* b_o = (const float*)d_in[17];
  const float* ln_g = (const float*)d_in[18]; const float* ln_b = (const float*)d_in[19];
  const float* W_out = (const float*)d_in[20]; const float* b_out = (const float*)d_in[21];

  int N = in_sizes[0] / 64;  // IN_DIM = 64
  int M = in_sizes[1];

  char* ws = (char*)d_ws;
  size_t off = 0;
  auto take = [&](size_t bytes) {
    char* p = ws + off;
    off = (off + bytes + 255) & ~(size_t)255;
    return p;
  };
  _Float16* Wa_cm   = (_Float16*)take(128 * 128 * 2);
  _Float16* Win_cm  = (_Float16*)take(128 * 64 * 2);
  _Float16* Wq_cm   = (_Float16*)take(128 * 128 * 2);
  _Float16* Wk_cm   = (_Float16*)take(128 * 128 * 2);
  _Float16* Wv_cm   = (_Float16*)take(128 * 128 * 2);
  _Float16* Wp_cm   = (_Float16*)take(128 * 128 * 2);
  _Float16* Wo_cm   = (_Float16*)take(128 * 128 * 2);
  _Float16* Wout_cm = (_Float16*)take(128 * 128 * 2);
  float*    fbuf    = (float*)take((size_t)N * 128 * 4);
  _Float16* f16buf  = (_Float16*)take((size_t)N * 128 * 2);
  _Float16* qf      = (_Float16*)take((size_t)N * 128 * 2);
  _Float16* kf      = (_Float16*)take((size_t)N * 128 * 2);
  _Float16* vf      = (_Float16*)take((size_t)N * 128 * 2);
  _Float16* o16     = (_Float16*)take((size_t)M * 128 * 2);

  // weight conversion (tiny)
  cvt_w<<<(128 * 128 + 255) / 256, 256, 0, stream>>>(W_a, Wa_cm, 128, 128);
  cvt_w<<<(64 * 128 + 255) / 256, 256, 0, stream>>>(W_in, Win_cm, 64, 128);
  cvt_w<<<(128 * 128 + 255) / 256, 256, 0, stream>>>(W_q, Wq_cm, 128, 128);
  cvt_w<<<(128 * 128 + 255) / 256, 256, 0, stream>>>(W_k, Wk_cm, 128, 128);
  cvt_w<<<(128 * 128 + 255) / 256, 256, 0, stream>>>(W_v, Wv_cm, 128, 128);
  cvt_w<<<(128 * 128 + 255) / 256, 256, 0, stream>>>(W_p, Wp_cm, 128, 128);
  cvt_w<<<(128 * 128 + 255) / 256, 256, 0, stream>>>(W_o, Wo_cm, 128, 128);
  cvt_w<<<(128 * 128 + 255) / 256, 256, 0, stream>>>(W_out, Wout_cm, 128, 128);

  int tilesN = (N + 15) / 16;
  k_proj_in<<<(tilesN + 7) / 8, 256, 16384, stream>>>(feats, Win_cm, b_in, fbuf, f16buf, N);
  k_proj_qkv<<<(tilesN + 7) / 8, 256, 98304, stream>>>(f16buf, Wq_cm, Wk_cm, Wv_cm,
                                                       b_q, b_k, b_v, qf, kf, vf, N);
  k_ppf<<<(M + 3) / 4, 128, 65536 + 4 * SCR3_PER_WAVE, stream>>>(
      ppfs, node_idx, group_idx, Wa_cm, b_a, Wp_cm, b_p, qf, kf, vf, o16, M);
  int tilesM = (M + 15) / 16;
  k_out<<<(tilesM + 3) / 4, 128, 65536 + 4 * SCR4_PER_WAVE, stream>>>(
      o16, Wo_cm, b_o, fbuf, node_idx, ln_g, ln_b, Wout_cm, b_out, (float*)d_out, M);
}
